// transformer_env_ARC_7121055776783
// MI455X (gfx1250) — matrix operand order repaired and checked by reading, not compiled, not run on hardware
//
#include <hip/hip_runtime.h>

// MI455X (gfx1250) fused transformer-prefix encoder.
// ~1.67 TFLOP of GEMMs -> bf16 WMMA (v_wmma_f32_16x16x32_bf16), f32 accum.
// Weights converted/transposed to K-major bf16 once per launch (L2-resident,
// ~25MB << 192MB L2). Activations are LDS-resident per (batch,prefix) tile.
// Big GEMMs use 2x2 register blocking: 4 WMMAs per (2 ds + 2 vmem) fragment
// loads, so the XDL pipe is not starved by LDS/L2 request rate.

typedef unsigned short u16;
typedef unsigned int   u32;
typedef __attribute__((ext_vector_type(4)))  u32    u32x4;
typedef __attribute__((ext_vector_type(16))) __bf16 v16bf;
typedef __attribute__((ext_vector_type(8)))  float  v8f;

#define B_   16
#define T_   64
#define D_   512
#define H_   8
#define HD_  64
#define F_   2048
#define L_   4
#define SCALE_ 0.125f   // 1/sqrt(HD)

// LDS pitches (avoid pathological bank strides, keep 16B row alignment)
#define DP   536   // u16 pitch for [64][512] activation arrays
#define VTP  72    // u16 pitch for V^T [512][64]
#define SCP  66    // f32 pitch for scores [64][64]
#define AP   520   // f32 pitch for f32 accumulators [64][512]

union BfFrag { u32x4 q[2]; v16bf v; u16 us[16]; };
union V8Pack { u32x4 q;    u16 us[8]; };

__device__ __forceinline__ u16 f2bf(float f) {
  union { float f; u32 u; } c; c.f = f;
  return (u16)((c.u + 0x7FFFu + ((c.u >> 16) & 1u)) >> 16);  // RNE
}
__device__ __forceinline__ float bf2f(u16 h) {
  union { float f; u32 u; } c; c.u = ((u32)h) << 16;
  return c.f;
}

__device__ __forceinline__ v8f wmma_bf16(v16bf a, v16bf b, v8f c) {
  return __builtin_amdgcn_wmma_f32_16x16x32_bf16(false, a, false, b, (short)0, c,
                                                 false, false);
}

// A fragment (16x32 bf16) from a row-major [rows][pitch] source.
// ISA layout: lane&15 = row; lanes<16 hold K {0..7,16..23}, lanes>=16 {8..15,24..31}.
__device__ __forceinline__ v16bf loadA(const u16* s, int pitch, int m0, int k0, int lane) {
  const int row = m0 + (lane & 15);
  const int kb  = k0 + ((lane >> 4) << 3);
  const u16* p  = s + (size_t)row * pitch + kb;
  BfFrag u;
  u.q[0] = *(const u32x4*)(p);
  u.q[1] = *(const u32x4*)(p + 16);
  return u.v;
}

// B fragment (32x16 bf16) from a K-major ("transposed") [N][ldk] source.
// ISA layout: lane&15 = column; lanes<16 hold K {0..7,16..23}, lanes>=16 {8..15,24..31}.
__device__ __forceinline__ v16bf loadB(const u16* s, int ldk, int n0, int k0, int lane) {
  const int n  = n0 + (lane & 15);
  const int kb = k0 + ((lane >> 4) << 3);
  const u16* p = s + (size_t)n * ldk + kb;
  BfFrag u;
  u.q[0] = *(const u32x4*)(p);
  u.q[1] = *(const u32x4*)(p + 16);
  return u.v;
}

// A fragment built from f32 (softmax probabilities), converted to bf16.
__device__ __forceinline__ v16bf loadA_f32(const float* s, int pitch, int m0, int k0, int lane) {
  const int row = m0 + (lane & 15);
  const int kb  = k0 + ((lane >> 4) << 3);
  const float* p = s + (size_t)row * pitch + kb;
  BfFrag u;
#pragma unroll
  for (int j = 0; j < 8; ++j) { u.us[j] = f2bf(p[j]); u.us[8 + j] = f2bf(p[16 + j]); }
  return u.v;
}

// 2x2-blocked GEMM inner loop: A rows {am0, am0+16}, B cols {bn0, bn0+16}.
// 4 WMMAs per k-step from 2 A + 2 B fragment loads (1 load per WMMA).
__device__ __forceinline__ void gemm2x2(const u16* __restrict__ A, int ap, int am0,
                                        const u16* __restrict__ Bw, int ldk, int bn0,
                                        int k0, int lane,
                                        v8f& c00, v8f& c01, v8f& c10, v8f& c11) {
#pragma unroll
  for (int kt = 0; kt < 16; ++kt) {
    const int k = k0 + kt * 32;
    const v16bf a0 = loadA(A, ap, am0,      k, lane);
    const v16bf a1 = loadA(A, ap, am0 + 16, k, lane);
    const v16bf b0 = loadB(Bw, ldk, bn0,      k, lane);
    const v16bf b1 = loadB(Bw, ldk, bn0 + 16, k, lane);
    c00 = wmma_bf16(a0, b0, c00);
    c01 = wmma_bf16(a0, b1, c01);
    c10 = wmma_bf16(a1, b0, c10);
    c11 = wmma_bf16(a1, b1, c11);
  }
}

// ---------------- weight transpose+convert: dstT[n][k] = bf16(src[k][n]) -----
__global__ void k_cvtT(const float* __restrict__ src, u16* __restrict__ dstT,
                       int K, int N) {
  const int idx = blockIdx.x * 256 + threadIdx.x;
  if (idx >= K * N) return;
  const int n = idx / K, k = idx - n * K;
  dstT[idx] = f2bf(src[(size_t)k * N + n]);
}

// ---------------- h0 = relu(x @ fc_w + fc_b), bf16 [B*T][512] ----------------
__global__ void k_h0(const float* __restrict__ x, const float* __restrict__ fw,
                     const float* __restrict__ fb, u16* __restrict__ h0) {
  const int idx = blockIdx.x * 256 + threadIdx.x;   // 1024*512
  const int row = idx >> 9, col = idx & 511;
  const float* xr = x + row * 128;
  float acc = fb[col];
#pragma unroll 4
  for (int k = 0; k < 128; ++k) acc += xr[k] * fw[k * D_ + col];
  h0[idx] = f2bf(fmaxf(acc, 0.f));
}

// ---------------- fused attention layer: QKV, masked softmax, PV, O-proj, LN1
__global__ __launch_bounds__(256, 1)
void k_attn(const u16* __restrict__ hin, size_t inPrefStride, u16* __restrict__ hbuf,
            const u16* __restrict__ wqkvT, const float* __restrict__ bqkv,
            const u16* __restrict__ woT,  const float* __restrict__ bo,
            const float* __restrict__ g1, const float* __restrict__ be1) {
  extern __shared__ char smem[];
  u16*  hS  = (u16*)smem;            // [64][DP]  input tile (residual)
  u16*  qS  = hS + 64 * DP;          // [64][DP]  Q, later overwritten by O
  u16*  kS  = qS + 64 * DP;          // [64][DP]  K
  u16*  vtS = kS + 64 * DP;          // [512][VTP] V^T
  float* scS  = (float*)(vtS + 512 * VTP);  // [64][SCP] scores/probs
  float* comm = scS + 64 * SCP;             // [64][8]   LN partials
  float* tmpF = (float*)kS;          // [64][AP] f32, aliases kS+vtS after attention

  const int tid = threadIdx.x, lane = tid & 31, wave = tid >> 5;
  const int lm = lane & 15, lh = (lane >> 4) << 3;
  const int b = blockIdx.x, pfx = blockIdx.y;
  const u16* src = hin + (size_t)pfx * inPrefStride + (size_t)b * (T_ * D_);
  u16* dst = hbuf + ((size_t)pfx * B_ + b) * (T_ * D_);

  // stage input tile -> LDS (4096 x 16B vectors, 16 per thread)
#pragma unroll
  for (int j = 0; j < 16; ++j) {
    const int idx = tid + 256 * j;
    const int row = idx >> 6, c8 = (idx & 63) << 3;
    *(u32x4*)(hS + row * DP + c8) = *(const u32x4*)(src + (size_t)row * D_ + c8);
  }
  __syncthreads();

  // ---- QKV: [64,1536] = h @ WqkvT^T ; 2x2 blocks: 96 items, 12 per wave
  {
    auto storeQKV = [&](v8f& a, int m0v, int ncol) {
      const float bias = bqkv[ncol];
      if (ncol < 1024) {                     // Q or K: row-major bf16
        u16* q = (ncol < 512) ? qS : kS;
        const int cc = ncol & 511;
#pragma unroll
        for (int r = 0; r < 8; ++r) q[(m0v + r) * DP + cc] = f2bf(a[r] + bias);
      } else {                               // V: D-fragment column == V^T row run
        V8Pack pk;
#pragma unroll
        for (int r = 0; r < 8; ++r) pk.us[r] = f2bf(a[r] + bias);
        *(u32x4*)(vtS + (ncol - 1024) * VTP + m0v) = pk.q;
      }
    };
    for (int it = wave; it < 96; it += 8) {
      const int am0 = (it & 1) * 32, bn0 = (it >> 1) * 32;
      __builtin_prefetch(wqkvT + (size_t)bn0 * D_, 0, 1);
      v8f c00 = {}, c01 = {}, c10 = {}, c11 = {};
      gemm2x2(hS, DP, am0, wqkvT, D_, bn0, 0, lane, c00, c01, c10, c11);
      storeQKV(c00, am0 + lh,      bn0 + lm);
      storeQKV(c01, am0 + lh,      bn0 + 16 + lm);
      storeQKV(c10, am0 + 16 + lh, bn0 + lm);
      storeQKV(c11, am0 + 16 + lh, bn0 + 16 + lm);
    }
  }
  __syncthreads();

  // ---- attention per head (waves cooperate; O overwrites Q in-place)
  for (int h = 0; h < H_; ++h) {
    // S = Q_h @ K_h^T : 16 tiles, 2 per wave, K-dim = HD = 64
#pragma unroll
    for (int s2 = 0; s2 < 2; ++s2) {
      const int t = wave * 2 + s2, mt = t >> 2, nt = t & 3;
      v8f acc = {};
#pragma unroll
      for (int kt = 0; kt < 2; ++kt)
        acc = wmma_bf16(loadA(qS, DP, mt * 16, h * HD_ + kt * 32, lane),
                        loadB(kS, DP, nt * 16, h * HD_ + kt * 32, lane), acc);
      const int ncol = nt * 16 + lm;
      const int m0   = mt * 16 + lh;
#pragma unroll
      for (int r = 0; r < 8; ++r) scS[(m0 + r) * SCP + ncol] = acc[r];
    }
    __syncthreads();
    // masked softmax over keys j<=pfx (one thread per query row)
    if (tid < T_) {
      float* row = scS + tid * SCP;
      float mx = -3.0e38f;
      for (int j = 0; j <= pfx; ++j) mx = fmaxf(mx, row[j] * SCALE_);
      float ssum = 0.f;
      for (int j = 0; j <= pfx; ++j) { float e = __expf(row[j] * SCALE_ - mx); row[j] = e; ssum += e; }
      const float inv = 1.f / ssum;
      for (int j = 0; j < T_; ++j) row[j] = (j <= pfx) ? row[j] * inv : 0.f;
    }
    __syncthreads();
    // O_h = P @ V_h : K-dim = 64 keys; B operand from V^T is contiguous
#pragma unroll
    for (int s2 = 0; s2 < 2; ++s2) {
      const int t = wave * 2 + s2, mt = t >> 2, nt = t & 3;
      v8f acc = {};
#pragma unroll
      for (int kt = 0; kt < 2; ++kt)
        acc = wmma_bf16(loadA_f32(scS, SCP, mt * 16, kt * 32, lane),
                        loadB(vtS, VTP, h * HD_ + nt * 16, kt * 32, lane), acc);
      const int dcol = h * HD_ + nt * 16 + lm;
      const int m0   = mt * 16 + lh;
#pragma unroll
      for (int r = 0; r < 8; ++r) qS[(m0 + r) * DP + dcol] = f2bf(acc[r]);
    }
    __syncthreads();
  }

  // ---- O-proj + bias + residual -> f32 tmp (aliases K/V^T LDS); 2x2 blocks
  {
    auto storeO = [&](v8f& a, int m0v, int ncol) {
      const float bias = bo[ncol];
#pragma unroll
      for (int r = 0; r < 8; ++r)
        tmpF[(m0v + r) * AP + ncol] = a[r] + bias + bf2f(hS[(m0v + r) * DP + ncol]);
    };
    for (int it = wave; it < 32; it += 8) {   // 2 mp x 16 np
      const int am0 = (it & 1) * 32, bn0 = (it >> 1) * 32;
      __builtin_prefetch(woT + (size_t)bn0 * D_, 0, 1);
      v8f c00 = {}, c01 = {}, c10 = {}, c11 = {};
      gemm2x2(qS, DP, am0, woT, D_, bn0, 0, lane, c00, c01, c10, c11);
      storeO(c00, am0 + lh,      bn0 + lm);
      storeO(c01, am0 + lh,      bn0 + 16 + lm);
      storeO(c10, am0 + 16 + lh, bn0 + lm);
      storeO(c11, am0 + 16 + lh, bn0 + 16 + lm);
    }
  }
  __syncthreads();

  // ---- LayerNorm1: 4 threads per row; packed b128 output stores
  {
    const int row = tid >> 2, sub = tid & 3, c0 = sub * 128;
    const float* rp = tmpF + row * AP;
    float s1 = 0.f, s2 = 0.f;
    for (int c = c0; c < c0 + 128; ++c) { const float v = rp[c]; s1 += v; s2 += v * v; }
    comm[row * 8 + sub] = s1; comm[row * 8 + 4 + sub] = s2;
    __syncthreads();
    s1 = comm[row * 8 + 0] + comm[row * 8 + 1] + comm[row * 8 + 2] + comm[row * 8 + 3];
    s2 = comm[row * 8 + 4] + comm[row * 8 + 5] + comm[row * 8 + 6] + comm[row * 8 + 7];
    const float mean = s1 * (1.f / 512.f);
    const float var  = s2 * (1.f / 512.f) - mean * mean;
    const float rstd = rsqrtf(var + 1e-5f);
    for (int c = c0; c < c0 + 128; c += 8) {
      V8Pack pk;
#pragma unroll
      for (int j = 0; j < 8; ++j)
        pk.us[j] = f2bf((rp[c + j] - mean) * rstd * g1[c + j] + be1[c + j]);
      *(u32x4*)(dst + (size_t)row * D_ + c) = pk.q;
    }
  }
}

// ---------------- fused FFN layer: FF1+ReLU (chunked over F), FF2 accum, LN2
__global__ __launch_bounds__(256, 1)
void k_ffn(u16* __restrict__ hbuf,
           const u16* __restrict__ w1T, const float* __restrict__ bff1,
           const u16* __restrict__ w2T, const float* __restrict__ bff2,
           const float* __restrict__ g2, const float* __restrict__ be2) {
  extern __shared__ char smem[];
  u16*  hS   = (u16*)smem;                  // [64][DP] input tile (residual)
  u16*  cS   = hS + 64 * DP;                // [64][DP] FF1 activation chunk (bf16)
  float* accS = (float*)(cS + 64 * DP);     // [64][AP] FF2 f32 accumulator
  float* comm = accS + 64 * AP;             // [64][8]

  const int tid = threadIdx.x, lane = tid & 31, wave = tid >> 5;
  const int lm = lane & 15, lh = (lane >> 4) << 3;
  const int b = blockIdx.x, pfx = blockIdx.y;
  u16* tile = hbuf + ((size_t)pfx * B_ + b) * (T_ * D_);

#pragma unroll
  for (int j = 0; j < 16; ++j) {
    const int idx = tid + 256 * j;
    const int row = idx >> 6, c8 = (idx & 63) << 3;
    *(u32x4*)(hS + row * DP + c8) = *(const u32x4*)(tile + (size_t)row * D_ + c8);
  }
  for (int idx = tid; idx < 64 * AP; idx += 256) accS[idx] = 0.f;
  __syncthreads();

  for (int ch = 0; ch < 4; ++ch) {          // F = 4 chunks of 512
    const int fb = ch * 512;
    // FF1 chunk: relu(h @ W1T^T + b1) -> bf16 cS ; 2x2 blocks
    {
      auto store1 = [&](v8f& a, int m0v, int fcol) {
        const float bias = bff1[fcol];
        const int cc = fcol - fb;
#pragma unroll
        for (int r = 0; r < 8; ++r)
          cS[(m0v + r) * DP + cc] = f2bf(fmaxf(a[r] + bias, 0.f));
      };
      for (int it = wave; it < 32; it += 8) {
        const int am0 = (it & 1) * 32, bn0 = fb + (it >> 1) * 32;
        __builtin_prefetch(w1T + (size_t)bn0 * D_, 0, 1);
        v8f c00 = {}, c01 = {}, c10 = {}, c11 = {};
        gemm2x2(hS, DP, am0, w1T, D_, bn0, 0, lane, c00, c01, c10, c11);
        store1(c00, am0 + lh,      bn0 + lm);
        store1(c01, am0 + lh,      bn0 + 16 + lm);
        store1(c10, am0 + 16 + lh, bn0 + lm);
        store1(c11, am0 + 16 + lh, bn0 + 16 + lm);
      }
    }
    __syncthreads();
    // FF2 partial: accS += chunk @ W2T^T (each wave owns fixed tiles -> no races)
    {
      auto ldAcc = [&](int m0v, int ncol) {
        v8f a;
#pragma unroll
        for (int r = 0; r < 8; ++r) a[r] = accS[(m0v + r) * AP + ncol];
        return a;
      };
      auto stAcc = [&](v8f& a, int m0v, int ncol) {
#pragma unroll
        for (int r = 0; r < 8; ++r) accS[(m0v + r) * AP + ncol] = a[r];
      };
      for (int it = wave; it < 32; it += 8) {
        const int am0 = (it & 1) * 32, bn0 = (it >> 1) * 32;
        v8f c00 = ldAcc(am0 + lh,      bn0 + lm);
        v8f c01 = ldAcc(am0 + lh,      bn0 + 16 + lm);
        v8f c10 = ldAcc(am0 + 16 + lh, bn0 + lm);
        v8f c11 = ldAcc(am0 + 16 + lh, bn0 + 16 + lm);
        gemm2x2(cS, DP, am0, w2T, F_, bn0, fb, lane, c00, c01, c10, c11);
        stAcc(c00, am0 + lh,      bn0 + lm);
        stAcc(c01, am0 + lh,      bn0 + 16 + lm);
        stAcc(c10, am0 + 16 + lh, bn0 + lm);
        stAcc(c11, am0 + 16 + lh, bn0 + 16 + lm);
      }
    }
    __syncthreads();
  }

  // LayerNorm2 over (accS + b2 + residual); packed b128 output stores
  {
    const int row = tid >> 2, sub = tid & 3, c0 = sub * 128;
    float s1 = 0.f, s2 = 0.f;
    for (int c = c0; c < c0 + 128; ++c) {
      const float v = accS[row * AP + c] + bff2[c] + bf2f(hS[row * DP + c]);
      s1 += v; s2 += v * v;
    }
    comm[row * 8 + sub] = s1; comm[row * 8 + 4 + sub] = s2;
    __syncthreads();
    s1 = comm[row * 8 + 0] + comm[row * 8 + 1] + comm[row * 8 + 2] + comm[row * 8 + 3];
    s2 = comm[row * 8 + 4] + comm[row * 8 + 5] + comm[row * 8 + 6] + comm[row * 8 + 7];
    const float mean = s1 * (1.f / 512.f);
    const float var  = s2 * (1.f / 512.f) - mean * mean;
    const float rstd = rsqrtf(var + 1e-5f);
    for (int c = c0; c < c0 + 128; c += 8) {
      V8Pack pk;
#pragma unroll
      for (int j = 0; j < 8; ++j) {
        const float v = accS[row * AP + c + j] + bff2[c + j] + bf2f(hS[row * DP + c + j]);
        pk.us[j] = f2bf((v - mean) * rstd * g2[c + j] + be2[c + j]);
      }
      *(u32x4*)(tile + (size_t)row * D_ + c) = pk.q;
    }
  }
}

// ---------------- gather prefix-diagonal + classifier + log_softmax ----------
__global__ __launch_bounds__(256)
void k_cls(const u16* __restrict__ hbuf, const float* __restrict__ w1,
           const float* __restrict__ b1, const float* __restrict__ w2,
           const float* __restrict__ b2, float* __restrict__ out) {
  __shared__ float hid[256];
  const int b = blockIdx.x, i = blockIdx.y, tid = threadIdx.x;
  const u16* y = hbuf + ((size_t)i * B_ + b) * (T_ * D_) + (size_t)i * D_;
  float acc = b1[tid];
  for (int k = 0; k < D_; ++k) acc += bf2f(y[k]) * w1[k * 256 + tid];
  hid[tid] = fmaxf(acc, 0.f);
  __syncthreads();
  if (tid == 0) {
    float z0 = b2[0], z1 = b2[1];
    for (int j = 0; j < 256; ++j) { z0 += hid[j] * w2[j * 2]; z1 += hid[j] * w2[j * 2 + 1]; }
    const float m = fmaxf(z0, z1);
    const float lse = m + __logf(__expf(z0 - m) + __expf(z1 - m));
    out[((size_t)b * T_ + i) * 2 + 0] = z0 - lse;
    out[((size_t)b * T_ + i) * 2 + 1] = z1 - lse;
  }
}

// ---------------------------------------------------------------------------
extern "C" void kernel_launch(void* const* d_in, const int* in_sizes, int n_in,
                              void* d_out, int out_size, void* d_ws, size_t ws_size,
                              hipStream_t stream) {
  (void)in_sizes; (void)n_in; (void)out_size; (void)ws_size;
  const float* x      = (const float*)d_in[0];
  const float* fc_w   = (const float*)d_in[1];
  const float* fc_b   = (const float*)d_in[2];
  const float* w_qkv  = (const float*)d_in[3];
  const float* b_qkv  = (const float*)d_in[4];
  const float* w_o    = (const float*)d_in[5];
  const float* b_o    = (const float*)d_in[6];
  const float* ln1_g  = (const float*)d_in[7];
  const float* ln1_b  = (const float*)d_in[8];
  const float* w_ff1  = (const float*)d_in[9];
  const float* b_ff1  = (const float*)d_in[10];
  const float* w_ff2  = (const float*)d_in[11];
  const float* b_ff2  = (const float*)d_in[12];
  const float* ln2_g  = (const float*)d_in[13];
  const float* ln2_b  = (const float*)d_in[14];
  const float* cls_w1 = (const float*)d_in[15];
  const float* cls_b1 = (const float*)d_in[16];
  const float* cls_w2 = (const float*)d_in[17];
  const float* cls_b2 = (const float*)d_in[18];

  // workspace layout (u16 units)
  u16* ws    = (u16*)d_ws;
  u16* qkvT  = ws;                                  // 4*1536*512
  u16* oT    = qkvT + (size_t)L_ * 1536 * 512;      // 4*512*512
  u16* ff1T  = oT   + (size_t)L_ * 512 * 512;       // 4*2048*512
  u16* ff2T  = ff1T + (size_t)L_ * 2048 * 512;      // 4*512*2048
  u16* h0    = ff2T + (size_t)L_ * 512 * 2048;      // 16*64*512
  u16* hbuf  = h0   + (size_t)B_ * T_ * D_;         // 64*16*64*512

  // transpose+convert weights to K-major bf16 (reused across all 64 prefixes)
  for (int l = 0; l < L_; ++l) {
    k_cvtT<<<(512 * 1536 + 255) / 256, 256, 0, stream>>>(
        w_qkv + (size_t)l * 512 * 1536, qkvT + (size_t)l * 1536 * 512, 512, 1536);
    k_cvtT<<<(512 * 512 + 255) / 256, 256, 0, stream>>>(
        w_o + (size_t)l * 512 * 512, oT + (size_t)l * 512 * 512, 512, 512);
    k_cvtT<<<(512 * 2048 + 255) / 256, 256, 0, stream>>>(
        w_ff1 + (size_t)l * 512 * 2048, ff1T + (size_t)l * 2048 * 512, 512, 2048);
    k_cvtT<<<(2048 * 512 + 255) / 256, 256, 0, stream>>>(
        w_ff2 + (size_t)l * 2048 * 512, ff2T + (size_t)l * 512 * 2048, 2048, 512);
  }

  k_h0<<<(B_ * T_ * D_) / 256, 256, 0, stream>>>(x, fc_w, fc_b, h0);

  constexpr unsigned SMEM_ATTN =
      (3u * 64 * DP + 512u * VTP) * 2 + (64u * SCP + 64u * 8) * 4;   // 298,496 B
  constexpr unsigned SMEM_FFN =
      (2u * 64 * DP) * 2 + (64u * AP + 64u * 8) * 4;                 // 272,384 B

  dim3 grid(B_, T_);   // (batch, prefix)
  for (int l = 0; l < L_; ++l) {
    const u16* hin = (l == 0) ? h0 : hbuf;
    const size_t ps = (l == 0) ? 0 : (size_t)B_ * T_ * D_;
    k_attn<<<grid, 256, SMEM_ATTN, stream>>>(
        hin, ps, hbuf,
        qkvT + (size_t)l * 1536 * 512, b_qkv + (size_t)l * 1536,
        oT + (size_t)l * 512 * 512,    b_o + (size_t)l * 512,
        ln1_g + (size_t)l * 512, ln1_b + (size_t)l * 512);
    k_ffn<<<grid, 256, SMEM_FFN, stream>>>(
        hbuf,
        ff1T + (size_t)l * 2048 * 512, b_ff1 + (size_t)l * 2048,
        ff2T + (size_t)l * 512 * 2048, b_ff2 + (size_t)l * 512,
        ln2_g + (size_t)l * 512, ln2_b + (size_t)l * 512);
  }

  k_cls<<<grid, 256, 0, stream>>>(hbuf, cls_w1, cls_b1, cls_w2, cls_b2, (float*)d_out);
}
